// TemporalSAGE_35287451304625
// MI455X (gfx1250) — matrix operand order repaired and checked by reading, not compiled, not run on hardware
//
#include <hip/hip_runtime.h>
#include <hip/hip_bf16.h>

typedef __bf16 bf16_t;
typedef __attribute__((ext_vector_type(16))) __bf16 v16bf;
typedef __attribute__((ext_vector_type(8)))  __bf16 v8bf;
typedef __attribute__((ext_vector_type(8)))  float  v8f;

#define IN_DIM   256
#define TIME_DIM 16
#define AUG      272
#define KPAD     288      // 272 padded to 9*32 for bf16 WMMA K-steps
#define HID      256
#define WROWS    512      // [W1_l ; W1_r] stacked

// ---------------------------------------------------------------------------
// Kernel 1: h_bf[n][0:288] = bf16( concat(x[n], time_embed[ts[n]], zeros16) )
// ---------------------------------------------------------------------------
__global__ __launch_bounds__(256)
void ts_prep_h(const float* __restrict__ x, const int* __restrict__ ts,
               const float* __restrict__ te, bf16_t* __restrict__ hb)
{
    const int n = blockIdx.x;
    const int t = threadIdx.x;
    bf16_t* row = hb + (size_t)n * KPAD;
    row[t] = (bf16_t)x[(size_t)n * IN_DIM + t];
    if (t < TIME_DIM) {
        const int tt = ts[n];
        row[IN_DIM + t]            = (bf16_t)te[tt * TIME_DIM + t];
        row[IN_DIM + TIME_DIM + t] = (bf16_t)0.0f;   // zero K padding
    }
}

// ---------------------------------------------------------------------------
// Kernel 2: Wcat[512][288] bf16 = [W1_l ; W1_r] with zero K padding
// ---------------------------------------------------------------------------
__global__ __launch_bounds__(KPAD)
void ts_prep_w(const float* __restrict__ wl, const float* __restrict__ wr,
               bf16_t* __restrict__ wc)
{
    const int j = blockIdx.x;      // 0..511 output channel
    const int k = threadIdx.x;     // 0..287
    float v = 0.0f;
    if (k < AUG) v = (j < HID) ? wl[j * AUG + k] : wr[(j - HID) * AUG + k];
    wc[(size_t)j * KPAD + k] = (bf16_t)v;
}

// ---------------------------------------------------------------------------
// Kernel 3: in-degree (float counts for the mean)
// ---------------------------------------------------------------------------
__global__ void ts_count(const int* __restrict__ dst, float* __restrict__ cnt, int E)
{
    const int e = blockIdx.x * blockDim.x + threadIdx.x;
    if (e < E) atomicAdd(&cnt[dst[e]], 1.0f);
}

// ---------------------------------------------------------------------------
// Kernel 4: WMMA GEMM  [z|y] = h_bf[N,288] @ Wcat[512,288]^T
//   z (cols 0..255)   -> bf16  (gather source for neighbor aggregation)
//   y (cols 256..511) -> f32   (self path, consumed by combine)
// One 16x16 D tile per wave32; 4 waves/WG cover 64 output cols.
// ---------------------------------------------------------------------------
__global__ __launch_bounds__(128)
void ts_gemm(const bf16_t* __restrict__ A,  // [N,288]
             const bf16_t* __restrict__ W,  // [512,288]
             bf16_t* __restrict__ Zb,       // [N,256]
             float*  __restrict__ Y,        // [N,256]
             int nrows)
{
    const int lane = threadIdx.x & 31;
    const int wid  = threadIdx.x >> 5;
    const int m0   = blockIdx.x * 16;
    const int j0   = blockIdx.y * 64 + wid * 16;
    const int g    = lane >> 4;        // lane half (ISA bf16 fragment layout)
    const int mn   = lane & 15;

    int arow = m0 + mn;  if (arow >= nrows) arow = nrows - 1;  // clamp, no EXEC change
    const int wrow = j0 + mn;                                   // output channel

    const bf16_t* aptr = A + (size_t)arow * KPAD;
    const bf16_t* bptr = W + (size_t)wrow * KPAD;

    v8f acc = {};
    #pragma unroll
    for (int kb = 0; kb < KPAD; kb += 32) {
        // A fragment (16x32 bf16): elems 0..15 at k = kb+16g .. kb+16g+15 (contiguous, matches B)
        v16bf a;
        {
            v8bf lo = *(const v8bf*)(aptr + kb + 16 * g);
            v8bf hi = *(const v8bf*)(aptr + kb + 16 * g + 8);
            #pragma unroll
            for (int i = 0; i < 8; ++i) { a[i] = lo[i]; a[8 + i] = hi[i]; }
        }
        // B fragment (32x16 bf16): 16 contiguous k values at k = kb+16g of the
        // row-major weight row (W row-major == B^T, exactly the operand WMMA wants)
        v16bf b = *(const v16bf*)(bptr + kb + 16 * g);

        acc = __builtin_amdgcn_wmma_f32_16x16x32_bf16(
                  false, a, false, b, (short)0, acc, false, false);
    }

    // D layout: lane -> col = j0 + (lane&15); VGPR v -> row = m0 + v + 8*(lane>>4)
    const int col = j0 + mn;
    if (col < HID) {
        #pragma unroll
        for (int v = 0; v < 8; ++v) {
            const int r = m0 + v + 8 * g;
            if (r < nrows) Zb[(size_t)r * HID + col] = (bf16_t)acc[v];
        }
    } else {
        const int yc = col - HID;
        #pragma unroll
        for (int v = 0; v < 8; ++v) {
            const int r = m0 + v + 8 * g;
            if (r < nrows) Y[(size_t)r * HID + yc] = acc[v];
        }
    }
}

// ---------------------------------------------------------------------------
// Kernel 5: edge scatter of z: sumz[dst] += z_bf[src]   (8 feats / thread)
// ---------------------------------------------------------------------------
__global__ __launch_bounds__(256)
void ts_scatter_z(const int* __restrict__ src, const int* __restrict__ dst,
                  const bf16_t* __restrict__ zb, float* __restrict__ sumz, int E)
{
    const long t = (long)blockIdx.x * blockDim.x + threadIdx.x;
    const int  e = (int)(t >> 5);
    if (e >= E) return;
    const int c = (int)(t & 31) * 8;
    const int s = src[e], d = dst[e];
    v8bf z = *(const v8bf*)(zb + (size_t)s * HID + c);
    float* op = sumz + (size_t)d * HID + c;
    #pragma unroll
    for (int i = 0; i < 8; ++i) atomicAdd(op + i, (float)z[i]);
}

// ---------------------------------------------------------------------------
// Kernel 6: combine — h1 = relu(sumz/deg + b1 + y); emit p = h1·Wh_l,
//           out = bh + h1·Wh_r. One node per 256-thread block (8 wave32).
// ---------------------------------------------------------------------------
__global__ __launch_bounds__(256)
void ts_combine(const float* __restrict__ sumz, const float* __restrict__ y,
                const float* __restrict__ cnt,  const float* __restrict__ b1,
                const float* __restrict__ whl,  const float* __restrict__ whr,
                const float* __restrict__ bh,
                float* __restrict__ pbuf, float* __restrict__ out)
{
    const int n = blockIdx.x;
    const int k = threadIdx.x;
    const float inv = 1.0f / fmaxf(cnt[n], 1.0f);
    float h1 = sumz[(size_t)n * HID + k] * inv + b1[k] + y[(size_t)n * HID + k];
    h1 = fmaxf(h1, 0.0f);
    float pv = h1 * whl[k];
    float qv = h1 * whr[k];
    #pragma unroll
    for (int off = 16; off > 0; off >>= 1) {     // wave32 butterfly
        pv += __shfl_xor(pv, off, 32);
        qv += __shfl_xor(qv, off, 32);
    }
    __shared__ float sp[8], sq[8];
    const int lane = k & 31, wv = k >> 5;
    if (lane == 0) { sp[wv] = pv; sq[wv] = qv; }
    __syncthreads();
    if (k == 0) {
        float P = 0.0f, Q = 0.0f;
        #pragma unroll
        for (int i = 0; i < 8; ++i) { P += sp[i]; Q += sq[i]; }
        pbuf[n] = P;
        out[n]  = bh[0] + Q;
    }
}

// ---------------------------------------------------------------------------
// Kernel 7: scalar edge scatter of p (the collapsed head aggregation)
// ---------------------------------------------------------------------------
__global__ void ts_scatter_p(const int* __restrict__ src, const int* __restrict__ dst,
                             const float* __restrict__ p, float* __restrict__ ssum, int E)
{
    const int e = blockIdx.x * blockDim.x + threadIdx.x;
    if (e < E) atomicAdd(&ssum[dst[e]], p[src[e]]);
}

// ---------------------------------------------------------------------------
// Kernel 8: out[n] += ssum[n]/deg
// ---------------------------------------------------------------------------
__global__ void ts_final(const float* __restrict__ ssum, const float* __restrict__ cnt,
                         float* __restrict__ out, int N)
{
    const int n = blockIdx.x * blockDim.x + threadIdx.x;
    if (n < N) out[n] += ssum[n] / fmaxf(cnt[n], 1.0f);
}

// ---------------------------------------------------------------------------
static inline size_t alignup(size_t v) { return (v + 255) & ~(size_t)255; }

extern "C" void kernel_launch(void* const* d_in, const int* in_sizes, int n_in,
                              void* d_out, int out_size, void* d_ws, size_t ws_size,
                              hipStream_t stream)
{
    const float* x    = (const float*)d_in[0];
    const int*   ei   = (const int*)  d_in[1];
    const int*   ts   = (const int*)  d_in[2];
    const float* te   = (const float*)d_in[3];
    const float* W1l  = (const float*)d_in[4];
    const float* b1l  = (const float*)d_in[5];
    const float* W1r  = (const float*)d_in[6];
    const float* Whl  = (const float*)d_in[7];
    const float* bhl  = (const float*)d_in[8];
    const float* Whr  = (const float*)d_in[9];
    float* out = (float*)d_out;

    const int N = in_sizes[0] / IN_DIM;
    const int E = in_sizes[1] / 2;
    const int* src = ei;
    const int* dst = ei + E;

    // workspace carve-up
    char* ws = (char*)d_ws;
    size_t off = 0;
    bf16_t* hb   = (bf16_t*)(ws + off); off = alignup(off + (size_t)N * KPAD * sizeof(bf16_t));
    bf16_t* wc   = (bf16_t*)(ws + off); off = alignup(off + (size_t)WROWS * KPAD * sizeof(bf16_t));
    bf16_t* zb   = (bf16_t*)(ws + off); off = alignup(off + (size_t)N * HID * sizeof(bf16_t));
    float*  y32  = (float*) (ws + off); off = alignup(off + (size_t)N * HID * sizeof(float));
    float*  sumz = (float*) (ws + off); off = alignup(off + (size_t)N * HID * sizeof(float));
    float*  cnt  = (float*) (ws + off); off = alignup(off + (size_t)N * sizeof(float));
    float*  pbuf = (float*) (ws + off); off = alignup(off + (size_t)N * sizeof(float));
    float*  ssum = (float*) (ws + off); off = alignup(off + (size_t)N * sizeof(float));

    // zero the accumulators every call (deterministic, graph-capturable)
    hipMemsetAsync(sumz, 0, (size_t)N * HID * sizeof(float), stream);
    hipMemsetAsync(cnt,  0, (size_t)N * sizeof(float), stream);
    hipMemsetAsync(ssum, 0, (size_t)N * sizeof(float), stream);

    ts_prep_h<<<N, 256, 0, stream>>>(x, ts, te, hb);
    ts_prep_w<<<WROWS, KPAD, 0, stream>>>(W1l, W1r, wc);
    ts_count<<<(E + 255) / 256, 256, 0, stream>>>(dst, cnt, E);

    dim3 ggrid((N + 15) / 16, WROWS / 64);
    ts_gemm<<<ggrid, 128, 0, stream>>>(hb, wc, zb, y32, N);

    long nt = (long)E * 32;
    ts_scatter_z<<<(unsigned)((nt + 255) / 256), 256, 0, stream>>>(src, dst, zb, sumz, E);

    ts_combine<<<N, 256, 0, stream>>>(sumz, y32, cnt, b1l, Whl, Whr, bhl, pbuf, out);
    ts_scatter_p<<<(E + 255) / 256, 256, 0, stream>>>(src, dst, pbuf, ssum, E);
    ts_final<<<(N + 255) / 256, 256, 0, stream>>>(ssum, cnt, out, N);
}
